// RNNRegressionModel_74062416053005
// MI455X (gfx1250) — compile-verified, matrix operand order checked by reading, not run on hardware
//
#include <hip/hip_runtime.h>
#include <math.h>

// Shapes: x:(64,512,2048) f32, H1=128, H2=64, F1=32, out:(64,1) f32
#define BATCH 64
#define TLEN  512
#define DIN   2048
#define H1    128
#define H2    64
#define F1    32
#define ROWS  (BATCH * TLEN)   // 32768

#define DEV __device__ __forceinline__

typedef __attribute__((ext_vector_type(16))) __bf16        v16bf;
typedef __attribute__((ext_vector_type(8)))  __bf16        v8bf;
typedef __attribute__((ext_vector_type(8)))  float         v8f;
typedef __attribute__((ext_vector_type(4)))  float         f4;
typedef __attribute__((ext_vector_type(8)))  unsigned int  u8x;

DEV unsigned short f2bf(float f) {            // round-half-up, 2 VALU
  unsigned int u = __builtin_bit_cast(unsigned int, f) + 0x8000u;
  return (unsigned short)(u >> 16);
}

// Pack bf16(f0) low half, bf16(f1) high half with one v_perm_b32.
DEV unsigned int pack_bf16x2(float f0, float f1) {
  unsigned int u0 = __builtin_bit_cast(unsigned int, f0) + 0x8000u;
  unsigned int u1 = __builtin_bit_cast(unsigned int, f1) + 0x8000u;
  return __builtin_amdgcn_perm(u1, u0, 0x07060302u);
}

DEV v8f wmma_bf16(v16bf a, v16bf b, v8f c) {
  return __builtin_amdgcn_wmma_f32_16x16x32_bf16(false, a, false, b, (short)0, c,
                                                 false, false);
}

// tanh(x) = 1 - 2/(e^{2x}+1): 2 trans + 3 VALU.
DEV float fast_tanh(float x) {
  float e = __expf(2.0f * x);
  return 1.0f - 2.0f * __builtin_amdgcn_rcpf(e + 1.0f);
}

// A-frag (16x32 MxK) or B-frag (32x16 KxN) from row-major K-contiguous bf16.
// Lane L<16 -> row L, K = {k0..k0+7, k0+16..k0+23}; lane L+16 -> K = {+8,+24}.
DEV v16bf load_frag_bf16(const unsigned short* base, long row0, int stride, int k0) {
  const int lane = threadIdx.x & 31, r = lane & 15, hi = lane >> 4;
  const unsigned short* p = base + (row0 + r) * (long)stride + k0 + hi * 8;
  v8bf lo = *(const v8bf*)p;
  v8bf hv = *(const v8bf*)(p + 16);
  return __builtin_shufflevector(lo, hv, 0,1,2,3,4,5,6,7,8,9,10,11,12,13,14,15);
}

// Same fragment from fp32 source (nontemporal stream + perm-packed conversion).
DEV v16bf load_frag_f32(const float* base, long row0, int stride, int k0) {
  const int lane = threadIdx.x & 31, r = lane & 15, hi = lane >> 4;
  const float* p = base + (row0 + r) * (long)stride + k0 + hi * 8;
  f4 c0 = __builtin_nontemporal_load((const f4*)p);
  f4 c1 = __builtin_nontemporal_load((const f4*)p + 1);
  f4 c2 = __builtin_nontemporal_load((const f4*)(p + 16));
  f4 c3 = __builtin_nontemporal_load((const f4*)(p + 16) + 1);
  u8x t;
  t[0] = pack_bf16x2(c0[0], c0[1]);
  t[1] = pack_bf16x2(c0[2], c0[3]);
  t[2] = pack_bf16x2(c1[0], c1[1]);
  t[3] = pack_bf16x2(c1[2], c1[3]);
  t[4] = pack_bf16x2(c2[0], c2[1]);
  t[5] = pack_bf16x2(c2[2], c2[3]);
  t[6] = pack_bf16x2(c3[0], c3[1]);
  t[7] = pack_bf16x2(c3[2], c3[3]);
  return __builtin_bit_cast(v16bf, t);
}

// ---------------------------------------------------------------- weight cvt
__global__ void k_f2bf(const float* __restrict__ src, unsigned short* __restrict__ dst,
                       int n) {
  int i = blockIdx.x * blockDim.x + threadIdx.x;
  if (i < n) dst[i] = f2bf(src[i]);
}

// ------------------------------------------- xp1 = x @ W_ih1^T + b_ih1 + b_hh1
// grid: ROWS/128 blocks x 256 threads (8 waves). Each wave: 16 rows x 128 cols.
// W k-slice (128x32 bf16 = 8KB) staged in LDS, double-buffered, shared by all
// 8 waves. launch_bounds(.,1) lifts the 128-VGPR occupancy cap (was spilling);
// sched_barrier keeps only one B-frag in flight per WMMA.
__global__ void __launch_bounds__(256, 1)
k_gemm1(const float* __restrict__ x, const unsigned short* __restrict__ Wbf,
        const float* __restrict__ b_ih, const float* __restrict__ b_hh,
        float* __restrict__ xp) {
  __shared__ __align__(16) unsigned short sW[2][H1 * 32];
  const int wave = threadIdx.x >> 5;
  const int lane = threadIdx.x & 31, r = lane & 15, hi = lane >> 4;
  const long row0 = (long)blockIdx.x * 128 + wave * 16;

  // staging: slice = 128 rows x 32 bf16 = 512 segments of 8 bf16; 2 per thread
  const int s0 = threadIdx.x, s1 = threadIdx.x + 256;
  const int r0s = s0 >> 2, p0s = (s0 & 3) * 8;
  const int r1s = s1 >> 2, p1s = (s1 & 3) * 8;

  v8f acc[8];
#pragma unroll
  for (int n = 0; n < 8; ++n) acc[n] = {};

  // preload slice 0
  v8bf st0 = *(const v8bf*)(Wbf + (long)r0s * DIN + p0s);
  v8bf st1 = *(const v8bf*)(Wbf + (long)r1s * DIN + p1s);
  *(v8bf*)&sW[0][r0s * 32 + p0s] = st0;
  *(v8bf*)&sW[0][r1s * 32 + p1s] = st1;
  __syncthreads();

  for (int i = 0; i < DIN / 32; ++i) {
    const int cur = i & 1, nxt = cur ^ 1;
    if (i + 1 < DIN / 32) {               // issue next-slice loads early
      const int kn = (i + 1) * 32;
      st0 = *(const v8bf*)(Wbf + (long)r0s * DIN + kn + p0s);
      st1 = *(const v8bf*)(Wbf + (long)r1s * DIN + kn + p1s);
    }
    v16bf a = load_frag_f32(x, row0, DIN, i * 32);
#pragma unroll
    for (int n = 0; n < 8; ++n) {
      v16bf b = load_frag_bf16(&sW[cur][0], n * 16, 32, 0);
      acc[n] = wmma_bf16(a, b, acc[n]);
      __builtin_amdgcn_sched_barrier(0);  // stop B-frag hoisting (reg pressure)
    }
    if (i + 1 < DIN / 32) {               // buf[nxt] was fully read last iter
      *(v8bf*)&sW[nxt][r0s * 32 + p0s] = st0;
      *(v8bf*)&sW[nxt][r1s * 32 + p1s] = st1;
    }
    __syncthreads();
  }

#pragma unroll
  for (int n = 0; n < 8; ++n) {
    const int col = n * 16 + r;
    const float bias = b_ih[col] + b_hh[col];
#pragma unroll
    for (int v = 0; v < 8; ++v) {
      const long row = row0 + v + 8 * hi;
      xp[row * H1 + col] = acc[n][v] + bias;
    }
  }
}

// ------------------------------ layer-1 recurrence: h = tanh(xp_t + h @ Whh^T)
// grid: 4 blocks x 256 threads (8 waves). Ping-pong h in LDS (1 barrier/step),
// W_hh B-frags hoisted to VGPRs, next step's xp prefetched into registers.
__global__ void __launch_bounds__(256, 1)
k_rnn1(const float* __restrict__ xp, const unsigned short* __restrict__ Whhbf,
       unsigned short* __restrict__ out1) {
  __shared__ __align__(16) unsigned short sh[2][16 * H1];
  const int wave = threadIdx.x >> 5;
  const int lane = threadIdx.x & 31, r = lane & 15, hi = lane >> 4;
  const int b0 = blockIdx.x * 16;
  const int n0 = wave * 16;
  const int col = n0 + r;
  for (int i = threadIdx.x; i < 16 * H1; i += blockDim.x) sh[0][i] = 0;  // h0 = 0
  v16bf bfrag[4];
#pragma unroll
  for (int kk = 0; kk < 4; ++kk) bfrag[kk] = load_frag_bf16(Whhbf, n0, H1, kk * 32);
  long idx0[8];
#pragma unroll
  for (int v = 0; v < 8; ++v)
    idx0[v] = (long)(b0 + v + 8 * hi) * TLEN * H1 + col;
  float xc[8];
#pragma unroll
  for (int v = 0; v < 8; ++v) xc[v] = xp[idx0[v]];      // t = 0
  __syncthreads();
  for (int t = 0; t < TLEN; ++t) {
    const int cur = t & 1, nxt = cur ^ 1;
    v16bf af[4];
#pragma unroll
    for (int kk = 0; kk < 4; ++kk) af[kk] = load_frag_bf16(sh[cur], 0, H1, kk * 32);
    const int tp = (t + 1 < TLEN) ? t + 1 : t;          // prefetch next xp slice
    float xn[8];
#pragma unroll
    for (int v = 0; v < 8; ++v) xn[v] = xp[idx0[v] + (long)tp * H1];
    v8f acc = {};
#pragma unroll
    for (int kk = 0; kk < 4; ++kk) acc = wmma_bf16(af[kk], bfrag[kk], acc);
#pragma unroll
    for (int v = 0; v < 8; ++v) {
      const int m = v + 8 * hi;
      const float val = fast_tanh(acc[v] + xc[v]);      // biases folded into xp
      const unsigned short bv = f2bf(val);
      out1[idx0[v] + (long)t * H1] = bv;
      sh[nxt][m * H1 + col] = bv;
      xc[v] = xn[v];
    }
    __syncthreads();
  }
}

// ------------------------------------- xp2 = out1 @ W_ih2^T + b_ih2 + b_hh2
__global__ void __launch_bounds__(128, 1)
k_gemm2(const unsigned short* __restrict__ a1, const unsigned short* __restrict__ Wbf,
        const float* __restrict__ b_ih, const float* __restrict__ b_hh,
        float* __restrict__ xp) {
  const int wave = threadIdx.x >> 5;
  const int lane = threadIdx.x & 31, r = lane & 15, hi = lane >> 4;
  const long row0 = (long)blockIdx.x * 64 + wave * 16;
  v8f acc[4];
#pragma unroll
  for (int n = 0; n < 4; ++n) acc[n] = {};
#pragma unroll
  for (int k0 = 0; k0 < H1; k0 += 32) {
    v16bf a = load_frag_bf16(a1, row0, H1, k0);
#pragma unroll
    for (int n = 0; n < 4; ++n) {
      v16bf b = load_frag_bf16(Wbf, n * 16, H1, k0);
      acc[n] = wmma_bf16(a, b, acc[n]);
    }
  }
#pragma unroll
  for (int n = 0; n < 4; ++n) {
    const int col = n * 16 + r;
    const float bias = b_ih[col] + b_hh[col];
#pragma unroll
    for (int v = 0; v < 8; ++v) {
      const long row = row0 + v + 8 * hi;
      xp[row * H2 + col] = acc[n][v] + bias;
    }
  }
}

// --------------------------- layer-2 recurrence; only final h (t=511) is kept
__global__ void __launch_bounds__(128, 1)
k_rnn2(const float* __restrict__ xp, const unsigned short* __restrict__ Whhbf,
       float* __restrict__ hlast) {
  __shared__ __align__(16) unsigned short sh[2][16 * H2];
  const int wave = threadIdx.x >> 5;
  const int lane = threadIdx.x & 31, r = lane & 15, hi = lane >> 4;
  const int b0 = blockIdx.x * 16;
  const int n0 = wave * 16;
  const int col = n0 + r;
  for (int i = threadIdx.x; i < 16 * H2; i += blockDim.x) sh[0][i] = 0;
  v16bf bfrag[2];
#pragma unroll
  for (int kk = 0; kk < 2; ++kk) bfrag[kk] = load_frag_bf16(Whhbf, n0, H2, kk * 32);
  long idx0[8];
#pragma unroll
  for (int v = 0; v < 8; ++v)
    idx0[v] = (long)(b0 + v + 8 * hi) * TLEN * H2 + col;
  float xc[8];
#pragma unroll
  for (int v = 0; v < 8; ++v) xc[v] = xp[idx0[v]];
  __syncthreads();
  for (int t = 0; t < TLEN; ++t) {
    const int cur = t & 1, nxt = cur ^ 1;
    v16bf af[2];
#pragma unroll
    for (int kk = 0; kk < 2; ++kk) af[kk] = load_frag_bf16(sh[cur], 0, H2, kk * 32);
    const int tp = (t + 1 < TLEN) ? t + 1 : t;
    float xn[8];
#pragma unroll
    for (int v = 0; v < 8; ++v) xn[v] = xp[idx0[v] + (long)tp * H2];
    v8f acc = {};
#pragma unroll
    for (int kk = 0; kk < 2; ++kk) acc = wmma_bf16(af[kk], bfrag[kk], acc);
#pragma unroll
    for (int v = 0; v < 8; ++v) {
      const int m = v + 8 * hi;
      const float val = fast_tanh(acc[v] + xc[v]);
      sh[nxt][m * H2 + col] = f2bf(val);
      if (t == TLEN - 1) hlast[(b0 + m) * H2 + col] = val;   // fp32 h_last
      xc[v] = xn[v];
    }
    __syncthreads();
  }
}

// --------------------- head: out = relu(h_last @ Wfc1^T + b1) @ Wfc2^T + b2
__global__ void __launch_bounds__(64, 1)
k_fc(const float* __restrict__ hlast, const float* __restrict__ Wfc1,
     const float* __restrict__ bfc1, const float* __restrict__ Wfc2,
     const float* __restrict__ bfc2, float* __restrict__ out) {
  const int b = threadIdx.x;
  if (b >= BATCH) return;
  float acc = bfc2[0];
#pragma unroll 4
  for (int j = 0; j < F1; ++j) {
    float s = bfc1[j];
    for (int k = 0; k < H2; ++k) s += hlast[b * H2 + k] * Wfc1[j * H2 + k];
    acc += fmaxf(s, 0.f) * Wfc2[j];
  }
  out[b] = acc;
}

extern "C" void kernel_launch(void* const* d_in, const int* in_sizes, int n_in,
                              void* d_out, int out_size, void* d_ws, size_t ws_size,
                              hipStream_t stream) {
  const float* x    = (const float*)d_in[0];
  const float* Wih1 = (const float*)d_in[1];
  const float* Whh1 = (const float*)d_in[2];
  const float* bih1 = (const float*)d_in[3];
  const float* bhh1 = (const float*)d_in[4];
  const float* Wih2 = (const float*)d_in[5];
  const float* Whh2 = (const float*)d_in[6];
  const float* bih2 = (const float*)d_in[7];
  const float* bhh2 = (const float*)d_in[8];
  const float* Wfc1 = (const float*)d_in[9];
  const float* bfc1 = (const float*)d_in[10];
  const float* Wfc2 = (const float*)d_in[11];
  const float* bfc2 = (const float*)d_in[12];
  float* out = (float*)d_out;

  // workspace layout (bytes, all 16B-aligned)
  char* ws = (char*)d_ws;
  unsigned short* Wih1b = (unsigned short*)(ws + 0);               // 128*2048*2
  unsigned short* Whh1b = (unsigned short*)(ws + 524288);          // 128*128*2
  unsigned short* Wih2b = (unsigned short*)(ws + 557056);          // 64*128*2
  unsigned short* Whh2b = (unsigned short*)(ws + 573440);          // 64*64*2
  float*          xp1   = (float*)(ws + 581632);                   // 32768*128*4
  unsigned short* out1b = (unsigned short*)(ws + 17358848);        // 32768*128*2
  float*          xp2   = (float*)(ws + 25747456);                 // 32768*64*4
  float*          hlast = (float*)(ws + 34136064);                 // 64*64*4

  k_f2bf<<<(H1 * DIN + 255) / 256, 256, 0, stream>>>(Wih1, Wih1b, H1 * DIN);
  k_f2bf<<<(H1 * H1 + 255) / 256, 256, 0, stream>>>(Whh1, Whh1b, H1 * H1);
  k_f2bf<<<(H2 * H1 + 255) / 256, 256, 0, stream>>>(Wih2, Wih2b, H2 * H1);
  k_f2bf<<<(H2 * H2 + 255) / 256, 256, 0, stream>>>(Whh2, Whh2b, H2 * H2);

  k_gemm1<<<ROWS / 128, 256, 0, stream>>>(x, Wih1b, bih1, bhh1, xp1);
  k_rnn1<<<4, 256, 0, stream>>>(xp1, Whh1b, out1b);
  k_gemm2<<<ROWS / 64, 128, 0, stream>>>(out1b, Wih2b, bih2, bhh2, xp2);
  k_rnn2<<<4, 128, 0, stream>>>(xp2, Whh2b, hlast);
  k_fc<<<1, 64, 0, stream>>>(hlast, Wfc1, bfc1, Wfc2, bfc2, out);
}
